// SCRFDPostModel_25314537242756
// MI455X (gfx1250) — matrix operand order repaired and checked by reading, not compiled, not run on hardware
//
#include <hip/hip_runtime.h>
#include <math.h>

typedef __attribute__((ext_vector_type(16))) _Float16 v16h;
typedef __attribute__((ext_vector_type(8)))  float    v8f;
typedef __attribute__((ext_vector_type(4)))  unsigned v4u;
typedef __attribute__((ext_vector_type(4)))  int      v4i;
typedef __attribute__((ext_vector_type(8)))  int      v8i;

#define NB    128
#define A8    12800
#define A16   3200
#define A32   800
#define N8    (NB * A8)          // 1638400
#define N16   (NB * A16)         // 409600
#define N32   (NB * A32)         // 102400
#define NTOT  (N8 + N16 + N32)   // 2150400
#define CHUNK 1024
#define NBLK  (NTOT / CHUNK)     // 2100 (chunks never straddle levels)
#define G8    (N8 / CHUNK)       // 1600
#define G16   (G8 + N16 / CHUNK) // 2000
#define NOBJ  100
#define NMS_IOU 0.45f
#define INFV  __builtin_inff()

// ---------------- helpers ----------------

__device__ __forceinline__ const float* cls_ptr(long gi, const float* c8,
                                                const float* c16, const float* c32) {
  if (gi < N8) return c8 + gi;
  gi -= N8;
  if (gi < N16) return c16 + gi;
  return c32 + (gi - N16);
}

__device__ __forceinline__ void emit_det(long gi, int rank,
                                         const float* b8, const float* b16, const float* b32,
                                         float rs0, float rs1, float score, float* det) {
  int b, a, stride, W;
  const float* bp;
  if (gi < N8) {
    b = (int)(gi / A8); a = (int)(gi % A8); stride = 8;  W = 80;
    bp = b8 + ((long)b * A8 + a) * 4;
  } else if (gi < (long)N8 + N16) {
    long r = gi - N8;
    b = (int)(r / A16); a = (int)(r % A16); stride = 16; W = 40;
    bp = b16 + ((long)b * A16 + a) * 4;
  } else {
    long r = gi - N8 - N16;
    b = (int)(r / A32); a = (int)(r % A32); stride = 32; W = 20;
    bp = b32 + ((long)b * A32 + a) * 4;
  }
  int cell = a >> 1;                 // NUM_ANCHORS == 2, same center for both
  float acx = (float)(cell % W) * (float)stride;
  float acy = (float)(cell / W) * (float)stride;
  float d0 = bp[0] * (float)stride, d1 = bp[1] * (float)stride;
  float d2 = bp[2] * (float)stride, d3 = bp[3] * (float)stride;
  float x1 = acx - d0, y1v = acy - d1, x2 = acx + d2, y2v = acy + d3;
  float* o = det + ((long)b * NOBJ + rank) * 5;
  o[0] = y1v * rs0; o[1] = x1 * rs1; o[2] = y2v * rs0; o[3] = x2 * rs1; o[4] = score;
}

// Uniform per-block level select (chunks are level-aligned).
__device__ __forceinline__ const float* chunk_base(int g, const float* c8,
                                                   const float* c16, const float* c32) {
  if (g < G8)  return c8 + (long)g * CHUNK;
  if (g < G16) return c16 + (long)(g - G8) * CHUNK;
  return c32 + (long)(g - G16) * CHUNK;
}

// 1-D Tensor Data Mover load: nelem f32 from gaddr into LDS at lds_off.
__device__ __forceinline__ void tdm_load_1d(unsigned lds_off, const void* gaddr,
                                            unsigned nelem) {
  unsigned long long ga = (unsigned long long)(uintptr_t)gaddr;
  v4u g0;
  g0[0] = 1u;                                          // count=1 (valid), user mode
  g0[1] = lds_off;                                     // lds_addr (bytes)
  g0[2] = (unsigned)(ga & 0xffffffffu);                // global_addr[31:0]
  g0[3] = (unsigned)((ga >> 32) & 0x01ffffffu)         // global_addr[56:32]
          | (2u << 30);                                // type=2 ("image")
  v8i g1;
  g1[0] = (int)(2u << 16);      // workgroup_mask=0, data_size=2 (4 bytes)
  g1[1] = (int)(nelem << 16);   // tensor_dim0[15:0] @ bits 63:48
  g1[2] = (int)(1u << 16);      // tensor_dim0[31:16]=0, tensor_dim1=1 @ bits 111:80
  g1[3] = (int)(nelem << 16);   // tensor_dim1[31:16]=0, tile_dim0 @ bits 127:112
  g1[4] = 0;                    // tile_dim1=0 (unused), tile_dim2=0
  g1[5] = (int)nelem;           // tensor_dim0_stride[31:0]
  g1[6] = 0;                    // tensor_dim0_stride[47:32]=0, tensor_dim1_stride lo=0
  g1[7] = 0;
  v4i gz = {0, 0, 0, 0};
#if defined(__clang_major__) && (__clang_major__ >= 23)
  v8i gz8 = {0, 0, 0, 0, 0, 0, 0, 0};
  __builtin_amdgcn_tensor_load_to_lds(g0, g1, gz, gz, gz8, 0);
#else
  __builtin_amdgcn_tensor_load_to_lds(g0, g1, gz, gz, 0);
#endif
}

// ---------------- K0: init det buffer + ratio_sum ----------------

__global__ void k_init(const float* __restrict__ origin, float* __restrict__ det,
                       float* __restrict__ ratio) {
  int idx = blockIdx.x * blockDim.x + threadIdx.x;
  if (idx < NB * NOBJ * 5) det[idx] = -1.0f;
  if (idx == 0) {
    float r0 = 0.f, r1 = 0.f;
    for (int b = 0; b < NB; ++b) {
      r0 += origin[2 * b + 0] * (1.0f / 640.0f);
      r1 += origin[2 * b + 1] * (1.0f / 640.0f);
    }
    ratio[0] = r0; ratio[1] = r1;
  }
}

// ---------------- K1: per-chunk positive counts ----------------

__global__ void k_count(const float* __restrict__ c8, const float* __restrict__ c16,
                        const float* __restrict__ c32, unsigned* __restrict__ partial) {
  __shared__ unsigned sd[256];
  int g = blockIdx.x, t = threadIdx.x;
  const float* bp = chunk_base(g, c8, c16, c32);
  float4 v = *(const float4*)(bp + t * 4);   // aligned b128 load
  unsigned cnt = (v.x > 0.f) + (v.y > 0.f) + (v.z > 0.f) + (v.w > 0.f);

  // gfx1250 speculative prefetch of the next chunk
  long pf = (long)g * CHUNK + (long)t * 4 + CHUNK;
  if (pf < NTOT) __builtin_prefetch(cls_ptr(pf, c8, c16, c32), 0, 1);

  sd[t] = cnt; __syncthreads();
  for (int off = 128; off > 0; off >>= 1) {
    if (t < off) sd[t] += sd[t + off];
    __syncthreads();
  }
  if (t == 0) partial[g] = sd[0];
}

// ---------------- K2: exclusive scan of 2100 partials (1 block) ----------------

__global__ void k_scan(const unsigned* __restrict__ partial, unsigned* __restrict__ offs) {
  __shared__ unsigned sd[256];
  int t = threadIdx.x;
  unsigned carry = 0;
  for (int base = 0; base < NBLK; base += 256) {
    unsigned v = (base + t < NBLK) ? partial[base + t] : 0u;
    sd[t] = v; __syncthreads();
    for (int d = 1; d < 256; d <<= 1) {
      unsigned x = sd[t];
      unsigned y = (t >= d) ? sd[t - d] : 0u;
      __syncthreads();
      sd[t] = x + y;
      __syncthreads();
    }
    if (base + t < NBLK) offs[base + t] = carry + sd[t] - v;  // exclusive
    unsigned tot = sd[255];
    __syncthreads();
    carry += tot;
  }
}

// ---------------- K3: emit winning detections into global-rank slots ----------------

__global__ void k_emit(const float* __restrict__ c8, const float* __restrict__ b8,
                       const float* __restrict__ c16, const float* __restrict__ b16,
                       const float* __restrict__ c32, const float* __restrict__ b32,
                       const unsigned* __restrict__ offs, const float* __restrict__ ratio,
                       float* __restrict__ det) {
  int g = blockIdx.x, t = threadIdx.x;
  unsigned base_off = offs[g];
  if (base_off >= NOBJ) return;  // uniform exit: this chunk cannot own a slot

  __shared__ unsigned sd[256];
  const float* bpv = chunk_base(g, c8, c16, c32);
  float4 v = *(const float4*)(bpv + t * 4);
  float cv[4] = {v.x, v.y, v.z, v.w};
  bool p[4];
  unsigned cnt = 0;
#pragma unroll
  for (int k = 0; k < 4; ++k) { p[k] = (cv[k] > 0.0f); cnt += p[k] ? 1u : 0u; }

  sd[t] = cnt; __syncthreads();
  for (int d = 1; d < 256; d <<= 1) {
    unsigned x = sd[t];
    unsigned y = (t >= d) ? sd[t - d] : 0u;
    __syncthreads();
    sd[t] = x + y;
    __syncthreads();
  }
  unsigned running = base_off + (sd[t] - cnt);  // exclusive local prefix + chunk offset
  float rs0 = ratio[0], rs1 = ratio[1];
  long base = (long)g * CHUNK + (long)t * 4;    // global concat index
#pragma unroll
  for (int k = 0; k < 4; ++k) {
    if (p[k]) {
      if (running < NOBJ) {
        float score = 1.0f / (1.0f + expf(-cv[k]));
        emit_det(base + k, (int)running, b8, b16, b32, rs0, rs1, score, det);
      }
      ++running;
    }
  }
}

// ---------------- K4: TDM tile load + stable sort + NMS + WMMA compaction ----------------

__global__ void __launch_bounds__(128) k_nms(const float* __restrict__ det,
                                             float* __restrict__ out) {
  int b = blockIdx.x, t = threadIdx.x;

  __shared__ float stage[NOBJ * 5];   // TDM destination tile (2000 B)
  __shared__ float oy1[NOBJ], ox1[NOBJ], oy2[NOBJ], ox2[NOBJ], os[NOBJ];
  __shared__ float sy1[NOBJ], sx1[NOBJ], sy2[NOBJ], sx2[NOBJ], ss[NOBJ];
  __shared__ int   keep[NOBJ];
  __shared__ float keepf[256];
  __shared__ float rsm[256];
  __shared__ float ny1[NOBJ], nx1[NOBJ], ny2[NOBJ], nx2[NOBJ], nss[NOBJ];

  // Wave 0 issues a 1-D Tensor Data Mover transfer of this image's det tile.
  if (t < 32) {
    if (t == 0) {
      tdm_load_1d((unsigned)(uintptr_t)(void*)stage,
                  det + (long)b * NOBJ * 5, NOBJ * 5);
    }
    __builtin_amdgcn_s_wait_tensorcnt(0);
  }
  __syncthreads();

  if (t < NOBJ) {
    const float* p = stage + t * 5;
    oy1[t] = p[0]; ox1[t] = p[1]; oy2[t] = p[2]; ox2[t] = p[3]; os[t] = p[4];
  }
  __syncthreads();

  // Stable descending rank sort (matches jnp.argsort(-s, stable))
  if (t < NOBJ) {
    float sj = os[t];
    int r = 0;
    for (int i = 0; i < NOBJ; ++i) {
      float si = os[i];
      r += ((si > sj) || (si == sj && i < t)) ? 1 : 0;
    }
    sy1[r] = oy1[t]; sx1[r] = ox1[t]; sy2[r] = oy2[t]; sx2[r] = ox2[t]; ss[r] = os[t];
  }
  if (t < NOBJ) keep[t] = 1;
  __syncthreads();

  // Greedy sequential suppression
  for (int i = 0; i < NOBJ - 1; ++i) {
    if (keep[i]) {  // uniform (LDS broadcast)
      if (t > i && t < NOBJ && keep[t]) {
        float ai = (sy2[i] - sy1[i]) * (sx2[i] - sx1[i]);
        float aj = (sy2[t] - sy1[t]) * (sx2[t] - sx1[t]);
        float ih = fminf(sy2[i], sy2[t]) - fmaxf(sy1[i], sy1[t]);
        float iw = fminf(sx2[i], sx2[t]) - fmaxf(sx1[i], sx1[t]);
        ih = fmaxf(ih, 0.0f); iw = fmaxf(iw, 0.0f);
        float inter = ih * iw;
        float uni = ai + aj - inter;
        float iou = (uni > 0.0f) ? (inter / uni) : 0.0f;
        if (iou > NMS_IOU) keep[t] = 0;
      }
    }
    __syncthreads();
  }

  // keep flags -> 16x16 f16 matrix; inclusive row scan via V x U on the WMMA unit
  keepf[t] = 0.0f; keepf[t + 128] = 0.0f;
  __syncthreads();
  if (t < NOBJ) keepf[t] = keep[t] ? 1.0f : 0.0f;
  __syncthreads();

  if (t < 32) {  // wave 0, EXEC all ones (uniform guard)
    int lane = t, M = lane & 15, hi = lane >> 4;
    v16h a, bb;
#pragma unroll
    for (int e = 0; e < 16; ++e) {
      // A operand, contiguous k order used here: K = hi*16 + e (matches B)
      int K = hi * 16 + e;
      float av = (K < 16) ? keepf[M * 16 + K] : 0.0f;
      a[e] = (_Float16)av;
      // B operand, same contiguous k order: K = hi*16 + e, N = lane&15
      int Kb = hi * 16 + e;
      bb[e] = (_Float16)((Kb < 16 && Kb <= M) ? 1.0f : 0.0f);  // U upper-tri ones
    }
    v8f c = {};
    c = __builtin_amdgcn_wmma_f32_16x16x32_f16(false, a, false, bb,
                                               (short)0, c, false, false);
#pragma unroll
    for (int g2 = 0; g2 < 8; ++g2) {
      int row = g2 + 8 * hi;
      rsm[row * 16 + (lane & 15)] = c[g2];  // RS[row][col] = row-wise inclusive scan
    }
  }
  __syncthreads();

  if (t < NOBJ) { ny1[t] = 0.f; nx1[t] = 0.f; ny2[t] = 0.f; nx2[t] = 0.f; nss[t] = 0.f; }
  __syncthreads();

  if (t < NOBJ && keep[t]) {
    int row = t >> 4;
    float incl = rsm[t];
    for (int r = 0; r < row; ++r) incl += rsm[r * 16 + 15];  // carries of prior rows
    int pos = (int)(incl + 0.5f) - 1;                        // exclusive position
    if (pos < NOBJ) {
      ny1[pos] = sy1[t]; nx1[pos] = sx1[t]; ny2[pos] = sy2[t]; nx2[pos] = sx2[t];
      nss[pos] = ss[t];
    }
  }
  __syncthreads();

  if (t < NOBJ) {
    float v0 = ny1[t], v1 = nx1[t], v2 = ny2[t], v3 = nx2[t], sv = nss[t];
    v0 = (v0 == -1.0f || v0 == 0.0f) ? INFV : v0;
    v1 = (v1 == -1.0f || v1 == 0.0f) ? INFV : v1;
    v2 = (v2 == -1.0f || v2 == 0.0f) ? INFV : v2;
    v3 = (v3 == -1.0f || v3 == 0.0f) ? INFV : v3;
    sv = (sv == -1.0f) ? INFV : sv;
    float* o = out + ((long)b * NOBJ + t) * 6;
    o[0] = v0; o[1] = v1; o[2] = v2; o[3] = v3; o[4] = sv; o[5] = 0.0f;
  }
}

// ---------------- launch ----------------

extern "C" void kernel_launch(void* const* d_in, const int* in_sizes, int n_in,
                              void* d_out, int out_size, void* d_ws, size_t ws_size,
                              hipStream_t stream) {
  const float* c8  = (const float*)d_in[0];
  const float* b8  = (const float*)d_in[1];
  const float* c16 = (const float*)d_in[2];
  const float* b16 = (const float*)d_in[3];
  const float* c32 = (const float*)d_in[4];
  const float* b32 = (const float*)d_in[5];
  const float* origin = (const float*)d_in[6];
  float* out = (float*)d_out;

  char* ws = (char*)d_ws;
  float*    det     = (float*)ws;                       // 128*100*5 f32 = 256000 B
  float*    ratio   = (float*)(ws + 256000);            // 2 f32
  unsigned* partial = (unsigned*)(ws + 256256);         // 2100 u32
  unsigned* offs    = (unsigned*)(ws + 256256 + NBLK * 4 + 64);

  k_init<<<(NB * NOBJ * 5 + 255) / 256, 256, 0, stream>>>(origin, det, ratio);
  k_count<<<NBLK, 256, 0, stream>>>(c8, c16, c32, partial);
  k_scan<<<1, 256, 0, stream>>>(partial, offs);
  k_emit<<<NBLK, 256, 0, stream>>>(c8, b8, c16, b16, c32, b32, offs, ratio, det);
  k_nms<<<NB, 128, 0, stream>>>(det, out);
}
